// CoAttention_66219805769893
// MI455X (gfx1250) — matrix operand order repaired and checked by reading, not compiled, not run on hardware
//
#include <hip/hip_runtime.h>
#include <hip/hip_bf16.h>
#include <math.h>

// ---------- CDNA5 WMMA plumbing (wave32, v_wmma_f32_16x16x32_bf16) ----------
typedef __attribute__((ext_vector_type(16))) __bf16 v16bf;
typedef __attribute__((ext_vector_type(8)))  float  v8f;

union AFrag {
  v16bf v;
  unsigned short s[16];
  unsigned u[8];
};

__device__ __forceinline__ unsigned short f2bf(float f) {
  unsigned u = __float_as_uint(f);
  u += 0x7FFFu + ((u >> 16) & 1u);   // round-to-nearest-even
  return (unsigned short)(u >> 16);
}
__device__ __forceinline__ float bf2f(unsigned short s) {
  return __uint_as_float(((unsigned)s) << 16);
}
__device__ __forceinline__ unsigned pack2(float lo, float hi) {
  return ((unsigned)f2bf(hi) << 16) | (unsigned)f2bf(lo);
}
__device__ __forceinline__ v8f wmma_bf16(AFrag a, AFrag b, v8f c) {
  return __builtin_amdgcn_wmma_f32_16x16x32_bf16(
      false, a.v, false, b.v, (short)0, c, false, false);
}
__device__ __forceinline__ float sigmoidf(float x) {
  return 1.0f / (1.0f + __expf(-x));
}

// Async global->LDS copy (CDNA5, tracked by ASYNCcnt).  lds_addr is the
// wave-relative LDS byte offset (low 32 bits of a generic LDS pointer,
// ISA 10.2), goff is a 32-bit byte offset against the uniform SGPR base.
__device__ __forceinline__ void async_load_b128(unsigned lds_addr, unsigned goff,
                                                const void* base) {
  asm volatile("global_load_async_to_lds_b128 %0, %1, %2"
               :: "v"(lds_addr), "v"(goff), "s"(base)
               : "memory");
}

// Problem constants
#define CB   64
#define CN1  256
#define CN2  64
#define CD1  2048
#define CD2  1024
#define CNH  8
#define CH   128

// ---------- f32 -> bf16 bulk convert (one float4 -> 4 bf16 per thread) ------
__global__ void cvt_bf16_kernel(const float* __restrict__ in,
                                unsigned short* __restrict__ out) {
  size_t i = (size_t)blockIdx.x * blockDim.x + threadIdx.x;
  float4 v = ((const float4*)in)[i];
  uint2 o;
  o.x = pack2(v.x, v.y);
  o.y = pack2(v.z, v.w);
  ((uint2*)out)[i] = o;
}

// ---------- Big NT GEMM, bf16 inputs, async double-buffered staging ----------
// C[M,N] = A[M,K] * B[N,K]^T + bias ; tile 128x64, K-step 32, 8 waves x 16x64.
#define BM 128
#define BN 64
#define BK 32

__global__ __launch_bounds__(256) void gemm_nt_bf16_kernel(
    const unsigned short* __restrict__ A, const unsigned short* __restrict__ Bm,
    const float* __restrict__ bias, const float* __restrict__ mul,
    float* __restrict__ outf, unsigned short* __restrict__ outb,
    int M, int N, int K, int mode) {
  __shared__ __align__(16) unsigned short As[2][BM][BK];  // 16 KB
  __shared__ __align__(16) unsigned short Bs[2][BN][BK];  // 8 KB
  const int tid = threadIdx.x;
  const int wave = tid >> 5, lane = tid & 31;
  const int lm = lane & 15, kh = lane >> 4;
  const int m0 = blockIdx.y * BM;
  const int n0 = blockIdx.x * BN;
  const int nk = K / BK;

  const int srow = tid >> 2;        // 0..63
  const int sc16 = (tid & 3) * 8;   // ushort column of a 16B chunk

  auto issue_stage = [&](int kb, int buf) {
    // A tile: 128 rows x 64B = 512 x b128 ; 2 per thread (3 async instr / wave)
#pragma unroll
    for (int r = 0; r < 2; ++r) {
      int row = srow + r * 64;
      unsigned lds = (unsigned)(size_t)&As[buf][row][sc16];
      unsigned go  = (unsigned)(((size_t)(m0 + row) * K + kb) * 2 + (size_t)(tid & 3) * 16);
      async_load_b128(lds, go, (const void*)A);
    }
    // B tile: 64 rows x 64B = 256 x b128 ; 1 per thread
    {
      unsigned lds = (unsigned)(size_t)&Bs[buf][srow][sc16];
      unsigned go  = (unsigned)(((size_t)(n0 + srow) * K + kb) * 2 + (size_t)(tid & 3) * 16);
      async_load_b128(lds, go, (const void*)Bm);
    }
  };

  issue_stage(0, 0);
  if (nk > 1) issue_stage(BK, 1);

  v8f acc[4] = {};
  for (int k = 0; k < nk; ++k) {
    const int buf = k & 1;
    // own stage-k transfers complete (3 of stage k+1 may stay in flight)
    if (k + 1 < nk) asm volatile("s_wait_asynccnt 3" ::: "memory");
    else            asm volatile("s_wait_asynccnt 0" ::: "memory");
    __syncthreads();  // whole tile (all waves' portions) visible

    AFrag a;
    {
      const unsigned* ar = (const unsigned*)&As[buf][wave * 16 + lm][0];
#pragma unroll
      for (int j = 0; j < 4; ++j) a.u[j]     = ar[kh * 4 + j];      // K = kh*8..
#pragma unroll
      for (int j = 0; j < 4; ++j) a.u[4 + j] = ar[8 + kh * 4 + j];  // K = 16+kh*8..
    }
#pragma unroll
    for (int nt = 0; nt < 4; ++nt) {
      AFrag b;
      const unsigned* brow = (const unsigned*)&Bs[buf][nt * 16 + lm][0];
#pragma unroll
      for (int j = 0; j < 4; ++j) b.u[j] = brow[kh * 4 + j];
#pragma unroll
      for (int j = 0; j < 4; ++j) b.u[4 + j] = brow[8 + kh * 4 + j];
      acc[nt] = wmma_bf16(a, b, acc[nt]);
    }
    __syncthreads();  // everyone done reading buf before it is refilled
    if (k + 2 < nk) issue_stage((k + 2) * BK, buf);
  }

  // C/D layout: vgpr j -> M = j + (lane/16)*8 ; lane%16 -> N
#pragma unroll
  for (int nt = 0; nt < 4; ++nt) {
    int col = n0 + nt * 16 + lm;
    float bv = bias[col];
#pragma unroll
    for (int j = 0; j < 8; ++j) {
      int row = m0 + wave * 16 + kh * 8 + j;
      float val = acc[nt][j] + bv;
      if (mode == 0) outb[(size_t)row * N + col] = f2bf(val);
      else outf[(size_t)row * N + col] = mul[(size_t)row * N + col] * sigmoidf(val);
    }
  }
}

// ---------- gates: gate1 = mean_n1 Q, gate2 = mean_n2 x2 ----------
__global__ void gates_kernel(const unsigned short* __restrict__ Qb,
                             const float* __restrict__ x2,
                             float* __restrict__ gate1, float* __restrict__ gate2) {
  int i = blockIdx.x * blockDim.x + threadIdx.x;  // 0 .. B*D2-1
  int b = i >> 10, o = i & 1023;
  float s1 = 0.f;
  for (int n = 0; n < CN1; ++n) s1 += bf2f(Qb[((size_t)(b * CN1 + n)) * CD2 + o]);
  float s2 = 0.f;
  for (int n = 0; n < CN2; ++n) s2 += x2[((size_t)(b * CN2 + n)) * CD2 + o];
  gate1[i] = s1 * (1.f / CN1);
  gate2[i] = s2 * (1.f / CN2);
}

// ---------- aff[b,h] = Q[b,:,h] * x2[b,:,h]^T / sqrt(H)  (256x64, K=128) ----
__global__ __launch_bounds__(256) void aff_kernel(const unsigned short* __restrict__ Qb,
                                                  const float* __restrict__ x2,
                                                  float* __restrict__ aff) {
  __shared__ unsigned Xs[64][64];  // x2 head slice as bf16: 64 rows x 128 K
  int bh = blockIdx.x, b = bh >> 3, h = bh & 7;
  const float* xb = x2 + (size_t)b * CN2 * CD2 + h * CH;
  for (int r = 0; r < 8; ++r) {
    int idx = threadIdx.x + r * 256;
    int row = idx >> 5, c4 = idx & 31;
    float4 v = *(const float4*)(xb + (size_t)row * CD2 + c4 * 4);
    Xs[row][c4 * 2 + 0] = pack2(v.x, v.y);
    Xs[row][c4 * 2 + 1] = pack2(v.z, v.w);
  }
  __syncthreads();
  int wave = threadIdx.x >> 5, lane = threadIdx.x & 31;
  int lm = lane & 15, kh = lane >> 4;
  const unsigned short* qb = Qb + (size_t)b * CN1 * CD2 + h * CH;
  v8f acc[2][4] = {};
  for (int kb = 0; kb < CH; kb += 32) {
    AFrag a[2];
    for (int mt = 0; mt < 2; ++mt) {
      int m = wave * 32 + mt * 16 + lm;
      const unsigned short* ap = qb + (size_t)m * CD2 + kb;
#pragma unroll
      for (int j = 0; j < 8; ++j) a[mt].s[j]     = ap[kh * 8 + j];
#pragma unroll
      for (int j = 0; j < 8; ++j) a[mt].s[8 + j] = ap[16 + kh * 8 + j];
    }
    for (int nt = 0; nt < 4; ++nt) {
      AFrag bf;
      int n = nt * 16 + lm;
#pragma unroll
      for (int j = 0; j < 4; ++j) bf.u[j] = Xs[n][(kb >> 1) + kh * 4 + j];
#pragma unroll
      for (int j = 0; j < 4; ++j) bf.u[4 + j] = Xs[n][(kb >> 1) + 8 + kh * 4 + j];
      acc[0][nt] = wmma_bf16(a[0], bf, acc[0][nt]);
      acc[1][nt] = wmma_bf16(a[1], bf, acc[1][nt]);
    }
  }
  const float scale = 0.08838834764831845f;  // 1/sqrt(128)
  float* ab = aff + (size_t)bh * CN1 * CN2;
  for (int mt = 0; mt < 2; ++mt)
    for (int nt = 0; nt < 4; ++nt)
#pragma unroll
      for (int j = 0; j < 8; ++j) {
        int row = wave * 32 + mt * 16 + kh * 8 + j;
        int col = nt * 16 + lm;
        ab[(size_t)row * CN2 + col] = acc[mt][nt][j] * scale;
      }
}

// ---------- softmax over N2 per n1-row; write v2q (transposed) + bf16 copy ---
__global__ void softmax_v2q_kernel(const float* __restrict__ aff,
                                   const unsigned char* __restrict__ q_mask,
                                   float* __restrict__ v2q_out,
                                   unsigned short* __restrict__ v2q_bf) {
  int bh = blockIdx.x, b = bh >> 3;
  int n1 = threadIdx.x;
  const float* row = aff + ((size_t)bh * CN1 + n1) * CN2;
  const unsigned char* qm = q_mask + (size_t)b * CN2;
  float mx = -1e30f;
  for (int j = 0; j < CN2; ++j) {
    float v = qm[j] ? -1e9f : row[j];
    mx = fmaxf(mx, v);
  }
  float s = 0.f;
  for (int j = 0; j < CN2; ++j) {
    float v = qm[j] ? -1e9f : row[j];
    s += __expf(v - mx);
  }
  float inv = 1.f / s;
  float* ob = v2q_out + (size_t)bh * CN2 * CN1;
  unsigned short* bb = v2q_bf + (size_t)bh * CN2 * CN1;
  for (int j = 0; j < CN2; ++j) {
    float v = qm[j] ? -1e9f : row[j];
    float p = __expf(v - mx) * inv;
    ob[(size_t)j * CN1 + n1] = p;   // [b,h,n2,n1]
    bb[(size_t)j * CN1 + n1] = f2bf(p);
  }
}

// ---------- softmax over N1 per n2-column; write q2v + bf16 copy ----------
__global__ void softmax_q2v_kernel(const float* __restrict__ aff,
                                   const unsigned char* __restrict__ v_mask,
                                   float* __restrict__ q2v_out,
                                   unsigned short* __restrict__ q2v_bf) {
  int bh = blockIdx.x, b = bh >> 3;
  int n2 = threadIdx.x;
  const float* ab = aff + (size_t)bh * CN1 * CN2;
  const unsigned char* vm = v_mask + (size_t)b * CN1;
  float mx = -1e30f;
  for (int i = 0; i < CN1; ++i) {
    float v = vm[i] ? -1e9f : ab[(size_t)i * CN2 + n2];
    mx = fmaxf(mx, v);
  }
  float s = 0.f;
  for (int i = 0; i < CN1; ++i) {
    float v = vm[i] ? -1e9f : ab[(size_t)i * CN2 + n2];
    s += __expf(v - mx);
  }
  float inv = 1.f / s;
  float* ob = q2v_out + (size_t)bh * CN1 * CN2;
  unsigned short* bb = q2v_bf + (size_t)bh * CN1 * CN2;
  for (int i = 0; i < CN1; ++i) {
    float v = vm[i] ? -1e9f : ab[(size_t)i * CN2 + n2];
    float p = __expf(v - mx) * inv;
    ob[(size_t)i * CN2 + n2] = p;   // [b,h,n1,n2]
    bb[(size_t)i * CN2 + n2] = f2bf(p);
  }
}

// ---------- x2_ = v2q[64x256] * (Q*gate2)[256x128] per (b,h) ----------
__global__ __launch_bounds__(256) void x2mm_kernel(
    const unsigned short* __restrict__ v2q_bf, const unsigned short* __restrict__ Qb,
    const float* __restrict__ gate2, float* __restrict__ x2cat) {
  int bh = blockIdx.x, b = bh >> 3, h = bh & 7;
  int wave = threadIdx.x >> 5, lane = threadIdx.x & 31;
  int lm = lane & 15, kh = lane >> 4;
  int mt = wave & 3, ng = wave >> 2;
  const unsigned short* Ab = v2q_bf + (size_t)bh * CN2 * CN1;
  const unsigned short* qb = Qb + (size_t)b * CN1 * CD2 + h * CH;
  const float* g2 = gate2 + (size_t)b * CD2 + h * CH;
  float g[4];
  int cols[4];
  for (int nt = 0; nt < 4; ++nt) {
    cols[nt] = ng * 64 + nt * 16 + lm;
    g[nt] = g2[cols[nt]];
  }
  v8f acc[4] = {};
  int m = mt * 16 + lm;
  for (int kb = 0; kb < CN1; kb += 32) {
    AFrag a;
    const unsigned short* ap = Ab + (size_t)m * CN1 + kb;
#pragma unroll
    for (int j = 0; j < 16; ++j) a.s[j] = ap[kh * 16 + j];
    for (int nt = 0; nt < 4; ++nt) {
      AFrag bf;
#pragma unroll
      for (int t = 0; t < 16; ++t)
        bf.s[t] = f2bf(bf2f(qb[(size_t)(kb + kh * 16 + t) * CD2 + cols[nt]]) * g[nt]);
      acc[nt] = wmma_bf16(a, bf, acc[nt]);
    }
  }
  float* ob = x2cat + (size_t)b * CN2 * CD2 + h * CH;
  for (int nt = 0; nt < 4; ++nt)
#pragma unroll
    for (int j = 0; j < 8; ++j) {
      int row = mt * 16 + kh * 8 + j;
      ob[(size_t)row * CD2 + cols[nt]] = acc[nt][j];
    }
}

// ---------- x1_ = q2v[256x64] * (x2*gate1)[64x128] per (b,h), bf16 out ------
__global__ __launch_bounds__(256) void x1mm_kernel(
    const unsigned short* __restrict__ q2v_bf, const float* __restrict__ x2,
    const float* __restrict__ gate1, unsigned short* __restrict__ x1cat_bf) {
  int bh = blockIdx.x, b = bh >> 3, h = bh & 7;
  int half = blockIdx.y;
  int wave = threadIdx.x >> 5, lane = threadIdx.x & 31;
  int lm = lane & 15, kh = lane >> 4;
  int m0 = half * 128 + wave * 16;
  const unsigned short* Ab = q2v_bf + (size_t)bh * CN1 * CN2;
  const float* xb = x2 + (size_t)b * CN2 * CD2 + h * CH;
  const float* g1 = gate1 + (size_t)b * CD2 + h * CH;
  v8f acc[8] = {};
  int m = m0 + lm;
  for (int kb = 0; kb < CN2; kb += 32) {
    AFrag a;
    const unsigned short* ap = Ab + (size_t)m * CN2 + kb;
#pragma unroll
    for (int j = 0; j < 16; ++j) a.s[j] = ap[kh * 16 + j];
    for (int nt = 0; nt < 8; ++nt) {
      int col = nt * 16 + lm;
      float gv = g1[col];
      AFrag bf;
#pragma unroll
      for (int t = 0; t < 16; ++t)
        bf.s[t] = f2bf(xb[(size_t)(kb + kh * 16 + t) * CD2 + col] * gv);
      acc[nt] = wmma_bf16(a, bf, acc[nt]);
    }
  }
  unsigned short* ob = x1cat_bf + (size_t)b * CN1 * CD2 + h * CH;
  for (int nt = 0; nt < 8; ++nt)
#pragma unroll
    for (int j = 0; j < 8; ++j) {
      int row = m0 + kh * 8 + j;
      ob[(size_t)row * CD2 + nt * 16 + lm] = f2bf(acc[nt][j]);
    }
}

// ---------- emb2 = x2 * sigmoid(x2_) ----------
__global__ void emb2_kernel(const float* __restrict__ x2,
                            const float* __restrict__ x2cat,
                            float* __restrict__ emb2) {
  int i = blockIdx.x * blockDim.x + threadIdx.x;
  emb2[i] = x2[i] * sigmoidf(x2cat[i]);
}

// ---------------------------------------------------------------------------
extern "C" void kernel_launch(void* const* d_in, const int* in_sizes, int n_in,
                              void* d_out, int out_size, void* d_ws, size_t ws_size,
                              hipStream_t stream) {
  (void)in_sizes; (void)n_in; (void)out_size; (void)ws_size;
  const float* x1 = (const float*)d_in[0];         // [64,256,2048]
  const float* x2 = (const float*)d_in[1];         // [64,64,1024]
  const unsigned char* v_mask = (const unsigned char*)d_in[2];  // [64,1,1,256]
  const unsigned char* q_mask = (const unsigned char*)d_in[3];  // [64,1,1,64]
  const float* Wq = (const float*)d_in[4];         // [8,128,2048] -> [1024,2048]
  const float* bq = (const float*)d_in[5];         // [1024]
  const float* Wr = (const float*)d_in[6];         // [2048,1024]
  const float* br = (const float*)d_in[7];         // [2048]

  float* emb1 = (float*)d_out;                     // 33554432
  float* emb2 = emb1 + (size_t)33554432;           // 4194304
  float* v2q  = emb2 + (size_t)4194304;            // 8388608
  float* q2v  = v2q  + (size_t)8388608;            // 8388608

  char* ws = (char*)d_ws;
  unsigned short* x1_bf   = (unsigned short*)ws;  ws += (size_t)33554432 * 2;
  unsigned short* Wq_bf   = (unsigned short*)ws;  ws += (size_t)2097152 * 2;
  unsigned short* Wr_bf   = (unsigned short*)ws;  ws += (size_t)2097152 * 2;
  unsigned short* Qcat_bf = (unsigned short*)ws;  ws += (size_t)16777216 * 2;
  float* gate1 = (float*)ws;                      ws += (size_t)65536 * 4;
  float* gate2 = (float*)ws;                      ws += (size_t)65536 * 4;
  float* aff   = (float*)ws;                      ws += (size_t)8388608 * 4;
  unsigned short* v2q_bf = (unsigned short*)ws;   ws += (size_t)8388608 * 2;
  unsigned short* q2v_bf = (unsigned short*)ws;   ws += (size_t)8388608 * 2;
  float* x2cat = (float*)ws;                      ws += (size_t)4194304 * 4;
  unsigned short* x1cat_bf = (unsigned short*)ws; ws += (size_t)16777216 * 2;

  // 0) one-shot bf16 conversion of the big GEMM operands
  cvt_bf16_kernel<<<33554432 / 4 / 256, 256, 0, stream>>>(x1, x1_bf);
  cvt_bf16_kernel<<<2097152 / 4 / 256, 256, 0, stream>>>(Wq, Wq_bf);
  cvt_bf16_kernel<<<2097152 / 4 / 256, 256, 0, stream>>>(Wr, Wr_bf);
  // 1) Q = x1 * Wq^T + bq   (M=16384, N=1024, K=2048), async-staged WMMA
  gemm_nt_bf16_kernel<<<dim3(1024 / BN, 16384 / BM), 256, 0, stream>>>(
      x1_bf, Wq_bf, bq, nullptr, nullptr, Qcat_bf, 16384, 1024, 2048, 0);
  // 2) gates
  gates_kernel<<<(CB * CD2) / 256, 256, 0, stream>>>(Qcat_bf, x2, gate1, gate2);
  // 3) affinity
  aff_kernel<<<CB * CNH, 256, 0, stream>>>(Qcat_bf, x2, aff);
  // 4,5) masked softmaxes (f32 outputs + bf16 probability copies)
  softmax_v2q_kernel<<<CB * CNH, CN1, 0, stream>>>(aff, q_mask, v2q, v2q_bf);
  softmax_q2v_kernel<<<CB * CNH, CN2, 0, stream>>>(aff, v_mask, q2v, q2v_bf);
  // 6) x2_ = v2q * (Q .* gate2)
  x2mm_kernel<<<CB * CNH, 256, 0, stream>>>(v2q_bf, Qcat_bf, gate2, x2cat);
  // 7) x1_ = q2v * (x2 .* gate1) -> bf16
  x1mm_kernel<<<dim3(CB * CNH, 2), 256, 0, stream>>>(q2v_bf, x2, gate1, x1cat_bf);
  // 8) emb1 = x1 * sigmoid(x1_ * Wr^T + br)  (M=16384, N=2048, K=1024)
  gemm_nt_bf16_kernel<<<dim3(2048 / BN, 16384 / BM), 256, 0, stream>>>(
      x1cat_bf, Wr_bf, br, x1, emb1, nullptr, 16384, 2048, 1024, 1);
  // 9) emb2 = x2 * sigmoid(x2_)
  emb2_kernel<<<4194304 / 256, 256, 0, stream>>>(x2, x2cat, emb2);
}
